// GAT_73521250173566
// MI455X (gfx1250) — matrix operand order repaired and checked by reading, not compiled, not run on hardware
//
#include <hip/hip_runtime.h>

// ---------------- problem constants (from reference) ----------------
#define AA 1024
#define CC 63        // nodes
#define DD 250
#define KTOT (AA*DD) // 256000
#define OUT 256
#define MPAD 64      // 63 nodes padded to 64 rows

// GEMM tiling
#define BK 512       // K per block
#define KT 32        // K per wmma step
#define NSTEP (BK/KT)
#define NBLK (KTOT/BK) // 500

typedef __attribute__((ext_vector_type(16))) __bf16 v16bf;
typedef __attribute__((ext_vector_type(2)))  __bf16 v2bf;
typedef __attribute__((ext_vector_type(8)))  float  v8f;
typedef __attribute__((ext_vector_type(4)))  int    i4;
typedef __attribute__((ext_vector_type(4)))  float  f4;
typedef __attribute__((ext_vector_type(2)))  float  f2;
// exact param type from the hipcc diagnostic for the async-LDS builtin:
typedef int v2i __attribute__((vector_size(2 * sizeof(int))));

union Frag { i4 q[2]; v16bf v; };

// ---- feature probes (degrade gracefully if a builtin is absent) ----
#if defined(__has_builtin)
#  if __has_builtin(__builtin_amdgcn_global_load_async_to_lds_b64) && \
      __has_builtin(__builtin_amdgcn_s_wait_asynccnt)
#    define USE_ASYNC_LDS 1
#  endif
#  if __has_builtin(__builtin_amdgcn_cvt_pk_bf16_f32)
#    define USE_CVT_PK 1
#  endif
#endif

#ifdef USE_ASYNC_LDS
#define GAS1(p) ((__attribute__((address_space(1))) v2i*)(p))
#define LAS3(p) ((__attribute__((address_space(3))) v2i*)(p))
#endif

static __device__ __forceinline__ unsigned short f2bf(float f) {
    unsigned u = __float_as_uint(f);
    u += 0x7FFFu + ((u >> 16) & 1u);   // round-to-nearest-even
    return (unsigned short)(u >> 16);
}

// pack two f32 -> packed bf16x2 (even k in [15:0])
static __device__ __forceinline__ unsigned pk(float lo, float hi) {
#ifdef USE_CVT_PK
    v2bf t = __builtin_amdgcn_cvt_pk_bf16_f32(lo, hi);
    return __builtin_bit_cast(unsigned, t);
#else
    return ((unsigned)f2bf(hi) << 16) | (unsigned)f2bf(lo);
#endif
}

// ---------------- kernel 0: zero the h accumulator ----------------
__global__ void gat_zero(float* __restrict__ h) {
    h[blockIdx.x * 256 + threadIdx.x] = 0.0f;
}

// ---------------- kernel 1: h = nodes @ W  (bf16 WMMA, K-split + atomics) ----
__global__ __launch_bounds__(256) void gat_gemm(const float* __restrict__ x,
                                                const float* __restrict__ W,
                                                float* __restrict__ h) {
    // LDS: A slab 64x32 f32 (async target), B slab col-major [256][32] bf16
    __shared__ __attribute__((aligned(16))) float          sAf[MPAD * KT]; // 8 KB
    __shared__ __attribute__((aligned(16))) unsigned short sB[OUT * KT];  // 16 KB

    const int tid   = threadIdx.x;
    const int lane  = tid & 31;
    const int w     = tid >> 5;     // wave 0..7
    const int mt    = w >> 1;       // M tile 0..3
    const int nhalf = w & 1;        // N half: cols [0,128) or [128,256)
    const int kblock = blockIdx.x * BK;

    // ---- per-thread invariant staging coordinates ----
    // W: thread owns cols c4..c4+3, rows kb4+4i (i=0..7): ONE base ptr + imm offsets
    const int c4  = (tid & 63) << 2;   // 0..252
    const int kb4 = tid >> 6;          // 0..3
    const float* wp = W + (size_t)(kblock + kb4) * OUT + c4;
    unsigned short* sBp = &sB[c4 * KT + kb4];
    // x: thread owns k-pair (2*kp, 2*kp+1), rows nb+16i (i=0..3)
    const int kp = tid & 15;           // k-pair index
    const int nb = tid >> 4;           // 0..15
    int a, d;                          // kg = a*250 + d, kg = kblock + 2*kp (+32/step)
    {
        int kg = kblock + (kp << 1);
        a = kg / DD;
        d = kg - a * DD;
    }

    // zero the pad row (row 63) once; async/fallback staging never writes it
    if (tid < 16) {
        f2 z; z.x = 0.0f; z.y = 0.0f;
        *(f2*)&sAf[(63 << 5) + (tid << 1)] = z;
    }

    v8f acc[8];
#pragma unroll
    for (int j = 0; j < 8; ++j)
#pragma unroll
        for (int r = 0; r < 8; ++r) acc[j][r] = 0.0f;

    for (int step = 0; step < NSTEP; ++step) {
        __syncthreads();   // previous step's fragment reads done before restaging

        // ---- stage x slab -> sAf (f32, rows 0..62; row 63 stays zero) ----
        const float* xp = x + ((size_t)a * CC + nb) * DD + d;  // row nb, this k-pair
#ifdef USE_ASYNC_LDS
#pragma unroll
        for (int i = 0; i < 3; ++i) {
            __builtin_amdgcn_global_load_async_to_lds_b64(
                GAS1(xp + i * 16 * DD),
                LAS3(&sAf[((nb + (i << 4)) << 5) + (kp << 1)]),
                0, 0);
        }
        if (nb != 15) {   // row nb+48 == 63 only for nb==15 (pad row)
            __builtin_amdgcn_global_load_async_to_lds_b64(
                GAS1(xp + 3 * 16 * DD),
                LAS3(&sAf[((nb + 48) << 5) + (kp << 1)]),
                0, 0);
        }
#else
#pragma unroll
        for (int i = 0; i < 3; ++i) {
            f2 xv = *(const f2*)(xp + i * 16 * DD);
            *(f2*)&sAf[((nb + (i << 4)) << 5) + (kp << 1)] = xv;
        }
        if (nb != 15) {
            f2 xv = *(const f2*)(xp + 3 * 16 * DD);
            *(f2*)&sAf[((nb + 48) << 5) + (kp << 1)] = xv;
        }
#endif

        // ---- stage W slab -> sB (cvt f32->bf16, col-major) ----
#pragma unroll
        for (int i = 0; i < 8; ++i) {
            const f4 wv = *(const f4*)(wp + (size_t)i * (4 * OUT));
            const unsigned p01 = pk(wv.x, wv.y);
            const unsigned p23 = pk(wv.z, wv.w);
            sBp[0 * KT + 4 * i] = (unsigned short)(p01 & 0xFFFFu);
            sBp[1 * KT + 4 * i] = (unsigned short)(p01 >> 16);
            sBp[2 * KT + 4 * i] = (unsigned short)(p23 & 0xFFFFu);
            sBp[3 * KT + 4 * i] = (unsigned short)(p23 >> 16);
        }
        // prefetch next step's W slab (one line per thread)
        if (step + 1 < NSTEP) __builtin_prefetch(wp + KT * OUT, 0, 0);

#ifdef USE_ASYNC_LDS
        __builtin_amdgcn_s_wait_asynccnt(0);
#endif
        __syncthreads();

        // advance streaming coordinates for next step
        wp += KT * OUT;
        d += KT;
        if (d >= DD) { d -= DD; ++a; }

        // ---- build A fragment (16x32): read f32 from LDS, cvt to bf16 ----
        Frag fa;
        {
            int m    = (mt << 4) + (lane & 15);
            int kofs = (lane >> 4) << 3;  // lanes 0-15: K 0..7/16..23; 16-31: 8..15/24..31
            const float* ap = &sAf[(m << 5) + kofs];
            f4 a0 = *(const f4*)(ap);
            f4 a1 = *(const f4*)(ap + 4);
            f4 a2 = *(const f4*)(ap + 16);
            f4 a3 = *(const f4*)(ap + 20);
            fa.q[0].x = pk(a0.x, a0.y); fa.q[0].y = pk(a0.z, a0.w);
            fa.q[0].z = pk(a1.x, a1.y); fa.q[0].w = pk(a1.z, a1.w);
            fa.q[1].x = pk(a2.x, a2.y); fa.q[1].y = pk(a2.z, a2.w);
            fa.q[1].z = pk(a3.x, a3.y); fa.q[1].w = pk(a3.z, a3.w);
        }
        // ---- 8 N-tiles: B fragment (32x16, bf16 straight from LDS) + WMMA ----
#pragma unroll
        for (int j = 0; j < 8; ++j) {
            Frag fb;
            int c  = (nhalf << 7) + (j << 4) + (lane & 15);
            int kb = (lane >> 4) << 3;    // lanes 0-15: K 0..7/16..23; lanes 16-31: K 8..15/24..31
            fb.q[0] = *(const i4*)&sB[(c << 5) + kb];
            fb.q[1] = *(const i4*)&sB[(c << 5) + kb + 16];
            acc[j] = __builtin_amdgcn_wmma_f32_16x16x32_bf16(
                false, fa.v, false, fb.v, (short)0, acc[j], false, false);
        }
    }

    // ---- writeback: C/D layout VGPR r: lanes0-15 (M=r,N=lane), 16-31 (M=r+8) ----
    {
        int mrow = (mt << 4) + ((lane >> 4) << 3);
        int ncol = (nhalf << 7) + (lane & 15);
#pragma unroll
        for (int j = 0; j < 8; ++j) {
#pragma unroll
            for (int r = 0; r < 8; ++r) {
                atomicAdd(&h[(mrow + r) * OUT + ncol + (j << 4)], acc[j][r]);
            }
        }
    }
}

// ---------------- kernel 2: attention + softmax + mean epilogue -------------
// out[n] = sum_j softmax_j(leaky(d[n]+s[j])) * mean_c(h[j]) + mean(bias)
__global__ void gat_finalize(const float* __restrict__ h,
                             const float* __restrict__ asrc,
                             const float* __restrict__ adst,
                             const float* __restrict__ bias,
                             float* __restrict__ out) {
    __shared__ float ss[64], sd[64], sm[64];
    __shared__ float bmean;
    const int t = threadIdx.x;
    if (t == 0) {
        float bs = 0.f;
        for (int c = 0; c < OUT; ++c) bs += bias[c];
        bmean = bs * (1.0f / OUT);
    }
    if (t < CC) {
        float s = 0.f, dv = 0.f, m = 0.f;
        for (int c = 0; c < OUT; ++c) {
            float hv = h[t * OUT + c];
            s += hv * asrc[c];
            dv += hv * adst[c];
            m += hv;
        }
        ss[t] = s; sd[t] = dv; sm[t] = m * (1.0f / OUT);
    }
    __syncthreads();
    if (t < CC) {
        const float dn = sd[t];
        float mx = -3.402823466e38f;
        for (int j = 0; j < CC; ++j) {
            float e = dn + ss[j];
            e = e > 0.f ? e : 0.2f * e;          // leaky_relu(0.2)
            mx = fmaxf(mx, e);
        }
        float se = 0.f, accv = 0.f;
        for (int j = 0; j < CC; ++j) {
            float e = dn + ss[j];
            e = e > 0.f ? e : 0.2f * e;
            float wj = __expf(e - mx);
            se += wj;
            accv += wj * sm[j];
        }
        out[t] = accv / se + bmean;
    }
}

// ---------------- launcher ----------------
extern "C" void kernel_launch(void* const* d_in, const int* in_sizes, int n_in,
                              void* d_out, int out_size, void* d_ws, size_t ws_size,
                              hipStream_t stream) {
    const float* x    = (const float*)d_in[0];  // (1024,1,63,250) f32
    const float* W    = (const float*)d_in[1];  // (256000,256) f32
    const float* asrc = (const float*)d_in[2];  // (256,) f32
    const float* adst = (const float*)d_in[3];  // (256,) f32
    const float* bias = (const float*)d_in[4];  // (256,) f32
    float* out = (float*)d_out;                 // 63 f32
    float* h   = (float*)d_ws;                  // 64x256 f32 accumulator

    gat_zero<<<MPAD, 256, 0, stream>>>(h);
    gat_gemm<<<NBLK, 256, 0, stream>>>(x, W, h);
    gat_finalize<<<1, 64, 0, stream>>>(h, asrc, adst, bias, out);
}
